// SoftPatternClassifier_63823214018795
// MI455X (gfx1250) — matrix operand order repaired and checked by reading, not compiled, not run on hardware
//
#include <hip/hip_runtime.h>

typedef __attribute__((ext_vector_type(16))) __bf16 v16bf;
typedef __attribute__((ext_vector_type(8)))  __bf16 v8bf;
typedef __attribute__((ext_vector_type(4)))  __bf16 v4bf;
typedef __attribute__((ext_vector_type(8)))  float  v8f;

// CDNA5 async global->LDS path (ASYNCcnt-tracked), with compile-safe fallback.
// Probe round 2 showed the builtin's params are pointers to a 16B int vector:
//   arg0: global (AS1 / __device__), arg1: LDS (AS3 / __shared__), imm, imm.
#if __has_builtin(__builtin_amdgcn_global_load_async_to_lds_b128)
#define SPN_ASYNC 1
typedef int spn_v4i __attribute__((vector_size(16)));
typedef __attribute__((address_space(1))) spn_v4i spn_gv4i;
typedef __attribute__((address_space(3))) spn_v4i spn_lv4i;
#else
#define SPN_ASYNC 0
#endif

#if __has_builtin(__builtin_amdgcn_s_wait_asynccnt)
#define SPN_WAIT_ASYNC() __builtin_amdgcn_s_wait_asynccnt(0)
#else
#define SPN_WAIT_ASYNC() asm volatile("s_wait_asynccnt 0x0" ::: "memory")
#endif

namespace {
constexpr int kB    = 64;     // batch
constexpr int kT    = 256;    // sequence length
constexpr int kD    = 300;    // embedding dim (K)
constexpr int kKpad = 320;    // K padded to 10 x 32
constexpr int kP    = 150;    // total patterns
constexpr int kR    = 1800;   // diag rows = P * 2 * 6
constexpr int kPPG  = 4;      // patterns per group
constexpr int kRPG  = 48;     // r columns per group = 3 WMMA N-tiles
constexpr int kGroups = 38;   // ceil(150/4)
constexpr int kRpad = kGroups * kRPG;  // 1824
constexpr int kM    = kB * kT;         // 16384
constexpr int kKS   = kKpad / 32;      // 10 k-steps
}

// ---------------------------------------------------------------------------
// Stage gathered embeddings -> bf16 hi/lo, row-major [M][Kpad], K zero-padded.
// ---------------------------------------------------------------------------
__global__ __launch_bounds__(256) void spn_stage_a(
    const int* __restrict__ docs, const float* __restrict__ emb,
    __bf16* __restrict__ Ahi, __bf16* __restrict__ Alo) {
  int idx = blockIdx.x * blockDim.x + threadIdx.x;
  const int total = kM * (kKpad / 4);
  if (idx >= total) return;
  int row = idx / (kKpad / 4);
  int k4  = (idx - row * (kKpad / 4)) * 4;
  int b   = row >> 8;
  int t   = row & 255;
  int tok = docs[b * kT + t];
  const float* src = emb + (size_t)tok * kD;
  v4bf hv, lv;
#pragma unroll
  for (int i = 0; i < 4; ++i) {
    int k = k4 + i;
    float x = (k < kD) ? src[k] : 0.0f;
    __bf16 h = (__bf16)x;
    hv[i] = h;
    lv[i] = (__bf16)(x - (float)h);
  }
  *(v4bf*)(Ahi + (size_t)row * kKpad + k4) = hv;
  *(v4bf*)(Alo + (size_t)row * kKpad + k4) = lv;
}

// ---------------------------------------------------------------------------
// Stage diags -> bf16 hi/lo, row-major [Rpad][Kpad], zero-padded both dims.
// ---------------------------------------------------------------------------
__global__ __launch_bounds__(256) void spn_stage_b(
    const float* __restrict__ diags,
    __bf16* __restrict__ Bhi, __bf16* __restrict__ Blo) {
  int idx = blockIdx.x * blockDim.x + threadIdx.x;
  const int total = kRpad * (kKpad / 4);
  if (idx >= total) return;
  int r  = idx / (kKpad / 4);
  int k4 = (idx - r * (kKpad / 4)) * 4;
  v4bf hv, lv;
#pragma unroll
  for (int i = 0; i < 4; ++i) {
    int k = k4 + i;
    float x = (r < kR && k < kD) ? diags[(size_t)r * kD + k] : 0.0f;
    __bf16 h = (__bf16)x;
    hv[i] = h;
    lv[i] = (__bf16)(x - (float)h);
  }
  *(v4bf*)(Bhi + (size_t)r * kKpad + k4) = hv;
  *(v4bf*)(Blo + (size_t)r * kKpad + k4) = lv;
}

// ---------------------------------------------------------------------------
// Fused split-bf16 WMMA GEMM (256 x 48 x 320 per block) + max-plus WFA scan.
// Block = 8 waves; block (g, b) owns batch b and patterns [4g, 4g+4).
// B tile is double-buffered in LDS (async global->LDS when available).
// ---------------------------------------------------------------------------
__global__ __launch_bounds__(256) void spn_gemm_scan(
    const __bf16* __restrict__ Ahi, const __bf16* __restrict__ Alo,
    const __bf16* __restrict__ Bhi, const __bf16* __restrict__ Blo,
    const float* __restrict__ bias, const float* __restrict__ epsilon,
    const int* __restrict__ doc_lens, float* __restrict__ scores) {
  __shared__ float ts[kT][kRPG];          // 49152 B: fused intermediate
  __shared__ uint4 bshq[2][2][kRPG][4];   // 12288 B: double-buffered hi/lo B tile

  const int g     = blockIdx.x;
  const int b     = blockIdx.y;
  const int tid   = threadIdx.x;
  const int wave  = tid >> 5;
  const int lane  = tid & 31;
  const int lhalf = lane >> 4;   // 0: lanes 0-15, 1: lanes 16-31
  const int l16   = lane & 15;
  const int rbase = g * kRPG;

  // B-stage chunk decomposition: 384 uint4 per k-step (2 mats x 48 rows x 4).
  auto gsrc = [&](int c, int kb) -> const uint4* {
    int mat = c / 192, rem = c - mat * 192, row = rem >> 2, q = rem & 3;
    return (const uint4*)((mat ? Blo : Bhi) + (size_t)(rbase + row) * kKpad + kb + q * 8);
  };
  auto ldst = [&](int c, int buf) -> uint4* {
    int mat = c / 192, rem = c - mat * 192, row = rem >> 2, q = rem & 3;
    return &bshq[buf][mat][row][q];
  };
#if SPN_ASYNC
  auto stage1 = [&](int c, int kb, int buf) {
    __builtin_amdgcn_global_load_async_to_lds_b128(
        (spn_gv4i*)(spn_v4i*)(void*)gsrc(c, kb),
        (spn_lv4i*)(spn_v4i*)(void*)ldst(c, buf), 0, 0);
  };
#endif

  v8f acc[2][3];
#pragma unroll
  for (int mi = 0; mi < 2; ++mi)
#pragma unroll
    for (int n = 0; n < 3; ++n)
      acc[mi][n] = (v8f){0.f, 0.f, 0.f, 0.f, 0.f, 0.f, 0.f, 0.f};

  // Prologue: stage k-step 0 into buffer 0.
#if SPN_ASYNC
  stage1(tid, 0, 0);
  if (tid < 128) stage1(tid + 256, 0, 0);
  SPN_WAIT_ASYNC();
#else
  {
    uint4 s0 = *gsrc(tid, 0);
    uint4 s1{};
    if (tid < 128) s1 = *gsrc(tid + 256, 0);
    *ldst(tid, 0) = s0;
    if (tid < 128) *ldst(tid + 256, 0) = s1;
  }
#endif
  __syncthreads();

  for (int ks = 0; ks < kKS; ++ks) {
    const int kb  = ks * 32;
    const int cur = ks & 1;
    const int nxt = cur ^ 1;
    // Issue next B tile fetch (overlaps with this step's WMMAs).
#if SPN_ASYNC
    if (ks + 1 < kKS) {
      stage1(tid, kb + 32, nxt);
      if (tid < 128) stage1(tid + 256, kb + 32, nxt);
    }
#else
    uint4 s0{}, s1{};
    if (ks + 1 < kKS) {
      s0 = *gsrc(tid, kb + 32);
      if (tid < 128) s1 = *gsrc(tid + 256, kb + 32);
    }
#endif
    // Per-lane A fragments (ISA 16-bit 16x32 A layout):
    //   elems 0..7  -> K = kb + lhalf*8  + i
    //   elems 8..15 -> K = kb + 16 + lhalf*8 + i
    v16bf ah[2], al[2];
#pragma unroll
    for (int mi = 0; mi < 2; ++mi) {
      const int t = (wave * 2 + mi) * 16 + l16;
      const size_t rowoff = (size_t)((b << 8) + t) * kKpad;
      v8bf h0 = *(const v8bf*)(Ahi + rowoff + kb + lhalf * 8);
      v8bf h1 = *(const v8bf*)(Ahi + rowoff + kb + 16 + lhalf * 8);
      v8bf l0 = *(const v8bf*)(Alo + rowoff + kb + lhalf * 8);
      v8bf l1 = *(const v8bf*)(Alo + rowoff + kb + 16 + lhalf * 8);
      ah[mi] = __builtin_shufflevector(h0, h1, 0,1,2,3,4,5,6,7,8,9,10,11,12,13,14,15);
      al[mi] = __builtin_shufflevector(l0, l1, 0,1,2,3,4,5,6,7,8,9,10,11,12,13,14,15);
    }
#pragma unroll
    for (int n = 0; n < 3; ++n) {
      // Per-lane B fragment: N = l16, elems 0..7 -> K = kb + lhalf*8 + i, elems 8..15 -> K = kb + 16 + lhalf*8 + i.
      const __bf16* bp0 = (const __bf16*)&bshq[cur][0][n * 16 + l16][0] + lhalf * 8;
      const __bf16* bp1 = (const __bf16*)&bshq[cur][1][n * 16 + l16][0] + lhalf * 8;
      v8bf bh0 = *(const v8bf*)bp0, bh1 = *(const v8bf*)(bp0 + 16);
      v8bf bl0 = *(const v8bf*)bp1, bl1 = *(const v8bf*)(bp1 + 16);
      v16bf bh = __builtin_shufflevector(bh0, bh1, 0,1,2,3,4,5,6,7,8,9,10,11,12,13,14,15);
      v16bf bl = __builtin_shufflevector(bl0, bl1, 0,1,2,3,4,5,6,7,8,9,10,11,12,13,14,15);
#pragma unroll
      for (int mi = 0; mi < 2; ++mi) {
        // Split-precision fp32 emulation: hi*hi + hi*lo + lo*hi.
        acc[mi][n] = __builtin_amdgcn_wmma_f32_16x16x32_bf16(
            false, ah[mi], false, bh, (short)0, acc[mi][n], false, false);
        acc[mi][n] = __builtin_amdgcn_wmma_f32_16x16x32_bf16(
            false, ah[mi], false, bl, (short)0, acc[mi][n], false, false);
        acc[mi][n] = __builtin_amdgcn_wmma_f32_16x16x32_bf16(
            false, al[mi], false, bh, (short)0, acc[mi][n], false, false);
      }
    }
    // Land next B tile in LDS; single barrier per k-step.
#if SPN_ASYNC
    SPN_WAIT_ASYNC();
#else
    if (ks + 1 < kKS) {
      *ldst(tid, nxt) = s0;
      if (tid < 128) *ldst(tid + 256, nxt) = s1;
    }
#endif
    __syncthreads();
  }

  // Scatter accumulators (+bias) into LDS ts tile per the 16x16 f32 C layout:
  // element j of v8f: M = j + lhalf*8, N = l16.
#pragma unroll
  for (int mi = 0; mi < 2; ++mi) {
    const int tb = (wave * 2 + mi) * 16;
#pragma unroll
    for (int n = 0; n < 3; ++n) {
      const int rl = n * 16 + l16;
      const int r  = rbase + rl;
      const float bv = (r < kR) ? bias[r] : 0.0f;
#pragma unroll
      for (int j = 0; j < 8; ++j)
        ts[tb + j + lhalf * 8][rl] = acc[mi][n][j] + bv;
    }
  }
  __syncthreads();

  // Max-plus WFA scan: 4 patterns per block, one chain per thread.
  if (tid < kPPG) {
    const int p = g * kPPG + tid;
    if (p < kP) {
      const int dlen = doc_lens[b];
      const int endi = (p < 50) ? 3 : (p < 100) ? 4 : 5;
      float eps[5];
#pragma unroll
      for (int i = 0; i < 5; ++i) eps[i] = epsilon[(size_t)p * 5 + i];
      float h[6] = {0.0f, -100.0f, -100.0f, -100.0f, -100.0f, -100.0f};
      float sc = -100.0f;
      const int rloc = tid * 12;
      for (int t = 0; t < kT; ++t) {
        const float* tm = &ts[t][rloc];  // [0..5]=self(d=0), [6..11]=main(d=1)
        float ae[6], hn[6];
        ae[0] = fmaxf(h[0], -100.0f);
#pragma unroll
        for (int l = 1; l < 6; ++l) ae[l] = fmaxf(h[l], h[l - 1] + eps[l - 1]);
        hn[0] = fmaxf(0.0f, ae[0] + tm[0]);
#pragma unroll
        for (int l = 1; l < 6; ++l)
          hn[l] = fmaxf(ae[l - 1] + tm[6 + (l - 1)], ae[l] + tm[l]);
        if (t < dlen) sc = fmaxf(sc, hn[endi]);
#pragma unroll
        for (int l = 0; l < 6; ++l) h[l] = hn[l];
      }
      scores[(size_t)b * kP + p] = sc;
    }
  }
}

// ---------------------------------------------------------------------------
// Tiny MLP head: out = relu(scores @ w1 + b1) @ w2 + b2   (64x150 -> 100 -> 2)
// ---------------------------------------------------------------------------
__global__ void spn_head(const float* __restrict__ scores,
                         const float* __restrict__ w1, const float* __restrict__ b1,
                         const float* __restrict__ w2, const float* __restrict__ b2,
                         float* __restrict__ out) {
  int b = blockIdx.x * blockDim.x + threadIdx.x;
  if (b >= kB) return;
  const float* s = scores + (size_t)b * kP;
  float o0 = b2[0], o1 = b2[1];
  for (int j = 0; j < 100; ++j) {
    float acc = b1[j];
    for (int p = 0; p < kP; ++p) acc = fmaf(s[p], w1[(size_t)p * 100 + j], acc);
    acc = fmaxf(acc, 0.0f);
    o0 = fmaf(acc, w2[j * 2 + 0], o0);
    o1 = fmaf(acc, w2[j * 2 + 1], o1);
  }
  out[b * 2 + 0] = o0;
  out[b * 2 + 1] = o1;
}

extern "C" void kernel_launch(void* const* d_in, const int* in_sizes, int n_in,
                              void* d_out, int out_size, void* d_ws, size_t ws_size,
                              hipStream_t stream) {
  (void)in_sizes; (void)n_in; (void)out_size; (void)ws_size;
  const int*   docs     = (const int*)d_in[0];
  const int*   doc_lens = (const int*)d_in[1];
  const float* emb      = (const float*)d_in[2];
  const float* diags    = (const float*)d_in[3];
  const float* bias     = (const float*)d_in[4];
  const float* eps      = (const float*)d_in[5];
  const float* w1       = (const float*)d_in[6];
  const float* b1       = (const float*)d_in[7];
  const float* w2       = (const float*)d_in[8];
  const float* b2       = (const float*)d_in[9];
  float* out = (float*)d_out;

  char* ws = (char*)d_ws;
  const size_t offA = (size_t)kM * kKpad * sizeof(__bf16);     // 10.5 MB
  const size_t offB = (size_t)kRpad * kKpad * sizeof(__bf16);  // 1.17 MB
  __bf16* Ahi = (__bf16*)ws;
  __bf16* Alo = (__bf16*)(ws + offA);
  __bf16* Bhi = (__bf16*)(ws + 2 * offA);
  __bf16* Blo = (__bf16*)(ws + 2 * offA + offB);
  float* scores = (float*)(ws + 2 * offA + 2 * offB);          // 64*150 f32

  const int nA = kM * (kKpad / 4);
  const int nB = kRpad * (kKpad / 4);
  spn_stage_a<<<(nA + 255) / 256, 256, 0, stream>>>(docs, emb, Ahi, Alo);
  spn_stage_b<<<(nB + 255) / 256, 256, 0, stream>>>(diags, Bhi, Blo);
  spn_gemm_scan<<<dim3(kGroups, kB), 256, 0, stream>>>(
      Ahi, Alo, Bhi, Blo, bias, eps, doc_lens, scores);
  spn_head<<<1, 64, 0, stream>>>(scores, w1, b1, w2, b2, out);
}
